// GCN_10187662426716
// MI455X (gfx1250) — matrix operand order repaired and checked by reading, not compiled, not run on hardware
//
#include <hip/hip_runtime.h>
#include <hip/hip_bf16.h>
#include <stdint.h>

// ---------------------------------------------------------------------------
// GCN layer: out = segment_sum(edge_val * (X@W)[edge_dst], edge_src) + bias
//
// MI455X plan:
//  * GEMM via v_wmma_f32_16x16x32_bf16 with bf16 hi/lo error compensation
//    (3 WMMAs per 16x16 tile -> ~fp32 accuracy, matrix-pipe throughput).
//  * W repacked once into per-fragment bf16 layout so B loads are coalesced.
//  * SpMM: 1 wave per edge; support (51MB) and out (51MB) are L2-resident
//    (192MB L2), scatter via native global_atomic_add_f32 (inline asm).
// ---------------------------------------------------------------------------

typedef __attribute__((ext_vector_type(16))) __bf16 v16bf;
typedef __attribute__((ext_vector_type(8)))  float  v8f;

#define IN_DIM   256
#define OUT_DIM  128
#define NTILES   8   // OUT_DIM / 16
#define KSTEPS   8   // IN_DIM / 32

// ---- native no-return FP32 global atomic add (CDNA5) ----------------------
__device__ __forceinline__ void gatomic_fadd(float* addr, float val) {
    asm volatile("global_atomic_add_f32 %0, %1, off"
                 :: "v"(addr), "v"(val) : "memory");
}

// ---------------------------------------------------------------------------
// Kernel 1: repack W [256,128] fp32 into fragment-ordered bf16 hi/lo arrays.
// Fragment index = ((ntile*KSTEPS + kstep)*32 + lane)*16 + e, where
// element e of the lane's v16bf holds W[K, N] with
//   K = (e<8 ? e : e+8) + (lane>=16 ? 8 : 0) + 32*kstep,  N = (lane&15) + 16*ntile
// matching the 16-bit B-matrix (32x16) VGPR layout.
// ---------------------------------------------------------------------------
__global__ void gcn_prep_w(const float* __restrict__ W,
                           __bf16* __restrict__ bhi,
                           __bf16* __restrict__ blo) {
    int idx = blockIdx.x * blockDim.x + threadIdx.x;   // 0 .. 32767
    if (idx >= NTILES * KSTEPS * 32 * 16) return;
    int e     = idx & 15;
    int lane  = (idx >> 4) & 31;
    int kstep = (idx >> 9) & 7;
    int ntile = (idx >> 12) & 7;
    int K = (e < 8 ? e : e + 8) + ((lane >= 16) ? 8 : 0) + 32 * kstep;
    int N = (lane & 15) + 16 * ntile;
    float w = W[K * OUT_DIM + N];
    __bf16 hi = (__bf16)w;
    float rem = w - (float)hi;
    bhi[idx] = hi;
    blo[idx] = (__bf16)rem;
}

// ---------------------------------------------------------------------------
// Kernel 2: support = X @ W via WMMA. One wave per 16-row strip.
// ---------------------------------------------------------------------------
__global__ void gcn_gemm(const float* __restrict__ X,
                         const __bf16* __restrict__ bhi,
                         const __bf16* __restrict__ blo,
                         float* __restrict__ support,
                         int nNodes, int nWaves) {
    int gtid = blockIdx.x * blockDim.x + threadIdx.x;
    int wave = gtid >> 5;
    int lane = threadIdx.x & 31;
    if (wave >= nWaves) return;               // wave-uniform: EXEC stays all-1s

    int row0 = wave << 4;
    int m    = lane & 15;                     // A-matrix M for this lane
    int row  = row0 + m;
    if (row >= nNodes) row = nNodes - 1;      // clamp loads; stores guarded
    const float* xrow = X + (size_t)row * IN_DIM;

    const int upper = (lane >= 16) ? 1 : 0;

    v8f acc[NTILES] = {};

    #pragma unroll
    for (int ks = 0; ks < KSTEPS; ++ks) {
        // ---- build A fragment (16-bit A 16x32 layout) in-register ----
        // element e -> K = (e<8 ? e : e+8) + (upper? 8:0) + 32*ks
        int k0 = ks * 32 + (upper ? 8 : 0);
        float4 a0 = *(const float4*)(xrow + k0);        // K = k0 .. k0+3
        float4 a1 = *(const float4*)(xrow + k0 + 4);    // K = k0+4 .. k0+7
        float4 a2 = *(const float4*)(xrow + k0 + 16);   // K = k0+16 .. k0+19
        float4 a3 = *(const float4*)(xrow + k0 + 20);   // K = k0+20 .. k0+23

        float av[16] = {a0.x,a0.y,a0.z,a0.w, a1.x,a1.y,a1.z,a1.w,
                        a2.x,a2.y,a2.z,a2.w, a3.x,a3.y,a3.z,a3.w};
        v16bf ahi, alo;
        #pragma unroll
        for (int e = 0; e < 16; ++e) {
            __bf16 h = (__bf16)av[e];
            ahi[e] = h;
            alo[e] = (__bf16)(av[e] - (float)h);
        }

        // ---- 8 N-tiles, 3 compensated WMMAs each ----
        #pragma unroll
        for (int t = 0; t < NTILES; ++t) {
            size_t fb = ((size_t)(t * KSTEPS + ks) * 32 + lane) * 16;
            v16bf wh = *(const v16bf*)(bhi + fb);   // 32B coalesced per lane
            v16bf wl = *(const v16bf*)(blo + fb);
            acc[t] = __builtin_amdgcn_wmma_f32_16x16x32_bf16(
                         false, ahi, false, wh, (short)0, acc[t], false, false);
            acc[t] = __builtin_amdgcn_wmma_f32_16x16x32_bf16(
                         false, ahi, false, wl, (short)0, acc[t], false, false);
            acc[t] = __builtin_amdgcn_wmma_f32_16x16x32_bf16(
                         false, alo, false, wh, (short)0, acc[t], false, false);
        }
    }

    // ---- store D: 32-bit C/D 16x16 layout: lane<16 -> M=r, lane>=16 -> M=r+8
    int ncol  = lane & 15;
    int moff  = upper ? 8 : 0;
    #pragma unroll
    for (int t = 0; t < NTILES; ++t) {
        #pragma unroll
        for (int r = 0; r < 8; ++r) {
            int rr = row0 + moff + r;
            if (rr < nNodes)
                support[(size_t)rr * OUT_DIM + t * 16 + ncol] = acc[t][r];
        }
    }
}

// ---------------------------------------------------------------------------
// Kernel 3: out[i,j] = bias[j]
// ---------------------------------------------------------------------------
__global__ void gcn_init_out(const float* __restrict__ bias,
                             float* __restrict__ out, int total) {
    int i = blockIdx.x * blockDim.x + threadIdx.x;
    if (i < total) out[i] = bias[i & (OUT_DIM - 1)];
}

// ---------------------------------------------------------------------------
// Kernel 4: one wave per edge; lane handles 4 consecutive output columns.
// out[src] += val * support[dst]   (native f32 atomics, L2-resident target)
// ---------------------------------------------------------------------------
__global__ void gcn_spmm(const float* __restrict__ support,
                         const int* __restrict__ esrc,
                         const int* __restrict__ edst,
                         const float* __restrict__ eval,
                         float* __restrict__ out, int nE) {
    int gtid = blockIdx.x * blockDim.x + threadIdx.x;
    int e    = gtid >> 5;
    int lane = threadIdx.x & 31;
    if (e >= nE) return;

    int   s = esrc[e];
    int   d = edst[e];
    float v = eval[e];

    const float4 g = *(const float4*)(support + (size_t)d * OUT_DIM + lane * 4);
    float* o = out + (size_t)s * OUT_DIM + lane * 4;
    gatomic_fadd(o + 0, g.x * v);
    gatomic_fadd(o + 1, g.y * v);
    gatomic_fadd(o + 2, g.z * v);
    gatomic_fadd(o + 3, g.w * v);
}

// ---------------------------------------------------------------------------
extern "C" void kernel_launch(void* const* d_in, const int* in_sizes, int n_in,
                              void* d_out, int out_size, void* d_ws, size_t ws_size,
                              hipStream_t stream) {
    const float* x      = (const float*)d_in[0];   // [N, 256]
    const float* weight = (const float*)d_in[1];   // [256, 128]
    const float* bias   = (const float*)d_in[2];   // [128]
    const int*   esrc   = (const int*)d_in[3];     // [E]
    const int*   edst   = (const int*)d_in[4];     // [E]
    const float* eval   = (const float*)d_in[5];   // [E]
    float*       out    = (float*)d_out;           // [N, 128]

    const int nNodes = in_sizes[0] / IN_DIM;       // 100000
    const int nE     = in_sizes[3];                // 1600000

    // workspace: [Bhi 64KB][Blo 64KB][support N*128*4B]
    __bf16* bhi     = (__bf16*)d_ws;
    __bf16* blo     = (__bf16*)((char*)d_ws + 65536);
    float*  support = (float*)((char*)d_ws + 131072);

    // 1) repack W fragments (32768 threads)
    gcn_prep_w<<<128, 256, 0, stream>>>(weight, bhi, blo);

    // 2) GEMM: one wave per 16-row strip
    int nWaves  = (nNodes + 15) / 16;              // 6250
    int threads = nWaves * 32;
    gcn_gemm<<<(threads + 255) / 256, 256, 0, stream>>>(x, bhi, blo, support,
                                                        nNodes, nWaves);

    // 3) out = bias
    int total = nNodes * OUT_DIM;
    gcn_init_out<<<(total + 255) / 256, 256, 0, stream>>>(bias, out, total);

    // 4) SpMM scatter-add: one wave per edge
    long long ethreads = (long long)nE * 32;
    gcn_spmm<<<(unsigned)((ethreads + 255) / 256), 256, 0, stream>>>(
        support, esrc, edst, eval, out, nE);
}
